// EmojiModel_26774826123829
// MI455X (gfx1250) — matrix operand order repaired and checked by reading, not compiled, not run on hardware
//
#include <hip/hip_runtime.h>

// ---------------------------------------------------------------------------
// EmojiModel (LSTM 2048-step, H=2048, IN=1024, E=512) for MI455X / gfx1250
// Strategy:
//   K1  convert x, W_ih, W_hh, W_dense to bf16 (RNE); init bias/h/barrier
//   K2  x_proj[2048][8192] = x·W_ihT + (b_ih+b_hh)       (WMMA bf16)
//   K3  persistent recurrence kernel, 1 global barrier/step (WMMA bf16 GEMV,
//       W_hh bf16 = 32MB resident in 192MB L2)
//   K4  fused dense GEMM (WMMA) + row softmax in LDS
// ---------------------------------------------------------------------------

#define SEQ 2048
#define IN  1024
#define H   2048
#define G   8192   // 4*H
#define E   512

typedef unsigned short u16;
typedef __attribute__((ext_vector_type(16))) __bf16       v16bf;
typedef __attribute__((ext_vector_type(8)))  float        v8f;
typedef __attribute__((ext_vector_type(4)))  unsigned int v4u;

union Frag16 {            // 16 bf16 = 32 bytes = two 128-bit loads
    v16bf v;
    v4u   q[2];
};

__device__ __forceinline__ u16 f2bf(float f) {
    unsigned u = __float_as_uint(f);
    u = (u + 0x7FFFu + ((u >> 16) & 1u)) >> 16;   // round-to-nearest-even
    return (u16)u;
}
__device__ __forceinline__ float sigf(float x) {
    return 1.0f / (1.0f + __expf(-x));
}

// --------------------------------------------------------------------------
// K1a: fp32 -> bf16 grid-stride convert
// --------------------------------------------------------------------------
__global__ void cvt_bf16_kernel(const float* __restrict__ src,
                                u16* __restrict__ dst, int n) {
    int i = blockIdx.x * blockDim.x + threadIdx.x;
    int stride = gridDim.x * blockDim.x;
    for (; i < n; i += stride) dst[i] = f2bf(src[i]);
}

// --------------------------------------------------------------------------
// K1b: bias_sum = b_ih + b_hh ; zero h double-buffer ; reset barrier
// --------------------------------------------------------------------------
__global__ void init_misc_kernel(const float* __restrict__ bih,
                                 const float* __restrict__ bhh,
                                 float* __restrict__ bias,
                                 u16* __restrict__ hbuf,
                                 unsigned* __restrict__ bar) {
    int i = blockIdx.x * blockDim.x + threadIdx.x;
    if (i < G)     bias[i] = bih[i] + bhh[i];
    if (i < 2 * H) hbuf[i] = 0;
    if (i < 2)     bar[i]  = 0u;
}

// --------------------------------------------------------------------------
// K2: x_proj[SEQ][G] = Xbf[SEQ][IN] * WihBf[G][IN]^T + bias
// one wave per 16x16 tile; 2048/16 * 8192/16 = 65536 waves
// --------------------------------------------------------------------------
__global__ void xproj_gemm_kernel(const u16* __restrict__ Xbf,
                                  const u16* __restrict__ Wbf,
                                  const float* __restrict__ bias,
                                  float* __restrict__ out) {
    int waveId = (blockIdx.x * blockDim.x + threadIdx.x) >> 5;
    int lane   = threadIdx.x & 31;
    int half   = lane >> 4;
    int l      = lane & 15;
    int tm     = waveId >> 9;        // G/16 = 512 column tiles
    int tn     = waveId & 511;
    int m0     = tm * 16;
    int n0     = tn * 16;

    v8f acc = {};
    for (int k0 = 0; k0 < IN; k0 += 32) {
        Frag16 a, b;
        const u16* ap = Xbf + (m0 + l) * IN + k0 + half * 8;
        a.q[0] = *(const v4u*)ap;
        a.q[1] = *(const v4u*)(ap + 16);
        const u16* bp = Wbf + (n0 + l) * IN + k0 + half * 8;
        b.q[0] = *(const v4u*)bp;
        b.q[1] = *(const v4u*)(bp + 16);
        acc = __builtin_amdgcn_wmma_f32_16x16x32_bf16(
            false, a.v, false, b.v, (short)0, acc, false, false);
    }
    float bv = bias[n0 + l];
    #pragma unroll
    for (int r = 0; r < 8; ++r)
        out[(size_t)(m0 + r + 8 * half) * G + n0 + l] = acc[r] + bv;
}

// --------------------------------------------------------------------------
// K3: persistent LSTM recurrence. 64 blocks x 256 threads (8 waves).
// Block b owns h slice j in [32b, 32b+32). Wave w: gate region r=w>>1,
// half-slice hh=w&1 -> computes gates n0..n0+15 with one WMMA chain
// (A = h broadcast into 16 rows; only D row 0 used). One grid barrier/step.
// --------------------------------------------------------------------------
__global__ void lstm_recurrence_kernel(const float* __restrict__ xproj,
                                       const u16* __restrict__ whhbf,
                                       u16* __restrict__ hbuf,   // [2][H]
                                       u16* __restrict__ hsbf,   // [SEQ][H]
                                       unsigned* __restrict__ bar) {
    __shared__ float gl[128];   // gates for this block's 32 h-elements (i,f,g,o)

    const int tid  = threadIdx.x;
    const int wave = tid >> 5;
    const int lane = tid & 31;
    const int half = lane >> 4;
    const int l    = lane & 15;
    const int b    = blockIdx.x;
    const int r    = wave >> 1;          // gate region 0..3 (i,f,g,o)
    const int hh   = wave & 1;
    const int n0   = r * H + b * 32 + hh * 16;

    float c_reg = 0.0f;                  // cell state, lives in registers

    for (int t = 0; t < SEQ; ++t) {
        const u16* hsrc = hbuf + (size_t)(t & 1) * H;
        u16*       hdst = hbuf + (size_t)((t + 1) & 1) * H;

        v8f acc = {};
        for (int k0 = 0; k0 < H; k0 += 32) {
            Frag16 a, bb;
            const u16* ap = hsrc + k0 + half * 8;          // broadcast loads
            a.q[0] = *(const v4u*)ap;
            a.q[1] = *(const v4u*)(ap + 16);
            const u16* bp = whhbf + (size_t)(n0 + l) * H + k0 + half * 8;
            bb.q[0] = *(const v4u*)bp;
            bb.q[1] = *(const v4u*)(bp + 16);
            acc = __builtin_amdgcn_wmma_f32_16x16x32_bf16(
                false, a.v, false, bb.v, (short)0, acc, false, false);
        }
        // D row 0 (M=0) lives in VGPR0 of lanes 0..15
        if (lane < 16)
            gl[r * 32 + hh * 16 + lane] =
                acc[0] + xproj[(size_t)t * G + n0 + lane];
        __syncthreads();

        if (tid < 32) {
            float iv = sigf(gl[tid]);
            float fv = sigf(gl[32 + tid]);
            float gv = tanhf(gl[64 + tid]);
            float ov = sigf(gl[96 + tid]);
            c_reg = fv * c_reg + iv * gv;
            float hn = ov * tanhf(c_reg);
            int j = b * 32 + tid;
            u16 hb = f2bf(hn);
            hdst[j] = hb;
            hsbf[(size_t)t * H + j] = hb;
        }
        __threadfence();
        __syncthreads();

        // ---- grid-wide barrier: monotone release counter ----
        if (tid == 0) {
            unsigned old = atomicAdd(&bar[0], 1u);
            if (old == (unsigned)(gridDim.x - 1)) {
                bar[0] = 0u;
                __threadfence();
                atomicAdd(&bar[1], 1u);          // release = t+1
            } else {
                while (atomicAdd(&bar[1], 0u) < (unsigned)(t + 1))
                    __builtin_amdgcn_s_sleep(2);
            }
        }
        __syncthreads();
    }
}

// --------------------------------------------------------------------------
// K4: logits[16 rows x 512] = hs_bf * W_dense^T + b_dense, then row softmax.
// One block per 16-row stripe; 8 waves x 4 tiles cover the 512 columns.
// Logits staged in 32KB LDS; softmax via wave shuffles (2 rows/wave).
// --------------------------------------------------------------------------
__global__ void dense_softmax_kernel(const u16* __restrict__ hsbf,
                                     const u16* __restrict__ wdbf,
                                     const float* __restrict__ bd,
                                     float* __restrict__ out) {
    __shared__ float lds[16 * E];
    const int wave = threadIdx.x >> 5;
    const int lane = threadIdx.x & 31;
    const int half = lane >> 4;
    const int l    = lane & 15;
    const int m0   = blockIdx.x * 16;

    for (int tt = 0; tt < 4; ++tt) {
        int n0 = wave * 64 + tt * 16;
        v8f acc = {};
        for (int k0 = 0; k0 < H; k0 += 32) {
            Frag16 a, b;
            const u16* ap = hsbf + (size_t)(m0 + l) * H + k0 + half * 8;
            a.q[0] = *(const v4u*)ap;
            a.q[1] = *(const v4u*)(ap + 16);
            const u16* bp = wdbf + (size_t)(n0 + l) * H + k0 + half * 8;
            b.q[0] = *(const v4u*)bp;
            b.q[1] = *(const v4u*)(bp + 16);
            acc = __builtin_amdgcn_wmma_f32_16x16x32_bf16(
                false, a.v, false, b.v, (short)0, acc, false, false);
        }
        float bv = bd[n0 + l];
        #pragma unroll
        for (int r = 0; r < 8; ++r)
            lds[(r + 8 * half) * E + n0 + l] = acc[r] + bv;
    }
    __syncthreads();

    // softmax: wave handles rows 2*wave, 2*wave+1 ; 16 elems/lane, stride 32
    for (int rr = 0; rr < 2; ++rr) {
        int row = wave * 2 + rr;
        float vals[16];
        float m = -3.4e38f;
        #pragma unroll
        for (int i = 0; i < 16; ++i) {
            vals[i] = lds[row * E + lane + i * 32];
            m = fmaxf(m, vals[i]);
        }
        for (int off = 16; off >= 1; off >>= 1)
            m = fmaxf(m, __shfl_xor(m, off, 32));
        float s = 0.0f;
        #pragma unroll
        for (int i = 0; i < 16; ++i) {
            vals[i] = __expf(vals[i] - m);
            s += vals[i];
        }
        for (int off = 16; off >= 1; off >>= 1)
            s += __shfl_xor(s, off, 32);
        float inv = 1.0f / s;
        #pragma unroll
        for (int i = 0; i < 16; ++i)
            out[(size_t)(m0 + row) * E + lane + i * 32] = vals[i] * inv;
    }
}

// --------------------------------------------------------------------------
// Host-side launch sequence (all on `stream`, graph-capture safe)
// --------------------------------------------------------------------------
extern "C" void kernel_launch(void* const* d_in, const int* in_sizes, int n_in,
                              void* d_out, int out_size, void* d_ws, size_t ws_size,
                              hipStream_t stream) {
    const float* x   = (const float*)d_in[0];   // [SEQ,1,IN]
    const float* Wih = (const float*)d_in[1];   // [G,IN]
    const float* Whh = (const float*)d_in[2];   // [G,H]
    const float* bih = (const float*)d_in[3];   // [G]
    const float* bhh = (const float*)d_in[4];   // [G]
    const float* Wd  = (const float*)d_in[5];   // [E,H]
    const float* bd  = (const float*)d_in[6];   // [E]
    float* out = (float*)d_out;                 // [SEQ, E] fp32 softmax

    char* ws = (char*)d_ws;
    // workspace layout (all offsets 256B-aligned by construction)
    size_t off = 0;
    u16*      xbf   = (u16*)     (ws + off); off += (size_t)SEQ * IN * 2;  //  4 MiB
    u16*      wihbf = (u16*)     (ws + off); off += (size_t)G * IN * 2;    // 16 MiB
    u16*      whhbf = (u16*)     (ws + off); off += (size_t)G * H  * 2;    // 32 MiB
    u16*      wdbf  = (u16*)     (ws + off); off += (size_t)E * H  * 2;    //  2 MiB
    float*    bias  = (float*)   (ws + off); off += (size_t)G * 4;         // 32 KiB
    float*    xproj = (float*)   (ws + off); off += (size_t)SEQ * G * 4;   // 64 MiB
    u16*      hsbf  = (u16*)     (ws + off); off += (size_t)SEQ * H * 2;   //  8 MiB
    u16*      hbuf  = (u16*)     (ws + off); off += (size_t)2 * H * 2;     //  8 KiB
    unsigned* bar   = (unsigned*)(ws + off); off += 256;

    // K1: precision conversion + state init
    cvt_bf16_kernel<<<512, 256, 0, stream>>>(x, xbf, SEQ * IN);
    cvt_bf16_kernel<<<2048, 256, 0, stream>>>(Wih, wihbf, G * IN);
    cvt_bf16_kernel<<<4096, 256, 0, stream>>>(Whh, whhbf, G * H);
    cvt_bf16_kernel<<<256, 256, 0, stream>>>(Wd, wdbf, E * H);
    init_misc_kernel<<<32, 256, 0, stream>>>(bih, bhh, bias, hbuf, bar);

    // K2: input projection GEMM -> x_proj (+ fused biases)
    // 65536 tiles / 8 waves per block = 8192 blocks
    xproj_gemm_kernel<<<8192, 256, 0, stream>>>(xbf, wihbf, bias, xproj);

    // K3: persistent recurrence, 64 co-resident blocks, grid barrier per step
    lstm_recurrence_kernel<<<64, 256, 0, stream>>>(xproj, whhbf, hbuf, hsbf, bar);

    // K4: fused dense GEMM + softmax, 128 blocks (16 rows each)
    dense_softmax_kernel<<<128, 256, 0, stream>>>(hsbf, wdbf, bd, out);
}
